// NGPTAttention_80169859547339
// MI455X (gfx1250) — matrix operand order repaired and checked by reading, not compiled, not run on hardware
//
#include <hip/hip_runtime.h>
#include <hip/hip_bf16.h>
#include <math.h>
#include <stdint.h>

// Problem constants (from reference)
#define HIDSZ 4096
#define NH    32
#define NKV   8
#define DHEAD 128
#define SEQ   2048
#define QSZ   (NH * DHEAD)          // 4096
#define KVSZ  (NKV * DHEAD)         // 1024
#define QKVN  (QSZ + 2 * KVSZ)      // 6144
#define GROUP (NH / NKV)            // 4

typedef __attribute__((ext_vector_type(16))) __bf16 v16bf;
typedef __attribute__((ext_vector_type(8)))  __bf16 v8bf;
typedef __attribute__((ext_vector_type(8)))  float  v8f;

static __device__ __forceinline__ v16bf cat8(v8bf lo, v8bf hi) {
  return __builtin_shufflevector(lo, hi, 0,1,2,3,4,5,6,7,8,9,10,11,12,13,14,15);
}

// CDNA5 async global->LDS copy, 16B per lane, tracked by ASYNCcnt.
// (inline asm: no builtin covers GLOBAL_LOAD_ASYNC_TO_LDS_*)
static __device__ __forceinline__ void async_ld16(uint32_t lds_addr, const void* gptr) {
  asm volatile("global_load_async_to_lds_b128 %0, %1, off"
               :: "v"(lds_addr), "v"((uint64_t)(uintptr_t)gptr)
               : "memory");
}
static __device__ __forceinline__ void wait_async0() {
  asm volatile("s_wait_asynccnt 0x0" ::: "memory");
}

// ---------------------------------------------------------------------------
// LDS-staged bf16 GEMM: C[M][N] (f32) = A[M][K] (bf16 rm) x B[K][N] (bf16 rm).
// Block = 256 threads (8 waves) computes a 128x128 C tile; K stepped by 32.
// Per step the block streams A(128x32) and B(32x128) into double-buffered LDS
// via global_load_async_to_lds_b128, drains with s_wait_asynccnt + barrier,
// and each wave runs 8 v_wmma_f32_16x16x32_bf16 on ds-loaded operands.
//
// CDNA5 operand layouts (ISA 7.12.2):
//  A 16x32 bf16 : half = lane>>4, m = lane&15; elems 0-7 = K 8*half..+7,
//                 elems 8-15 = K 16+8*half..+7.
//  B 32x16 bf16 : n = lane&15; elems 0-7 = K 8*half..+7,
//                 elems 8-15 = K 16+8*half..+7.
//  C/D 16x16 f32: elem i <-> row (i + 8*half), column = lane&15.
// ---------------------------------------------------------------------------
__global__ __launch_bounds__(256) void gemm_bf16_lds_kernel(
    const __bf16* __restrict__ A, const __bf16* __restrict__ B,
    float* __restrict__ C, int M, int N, int K) {
  __shared__ __align__(16) __bf16 As[2][128 * 32];   // [row][k] 8KB per buf
  __shared__ __align__(16) __bf16 Bs[2][32 * 128];   // [k][col] 8KB per buf

  const int tid  = threadIdx.x;
  const int lane = tid & 31;
  const int w    = tid >> 5;
  const int wm   = w >> 1;            // 0..3 : 32-row slice
  const int wn   = w & 1;             // 0..1 : 64-col slice
  const int half = lane >> 4;
  const int l16  = lane & 15;
  const int nb = N >> 7;
  const int m0 = (blockIdx.x / nb) * 128;
  const int n0 = (blockIdx.x % nb) * 128;

  // staging: 512 16B chunks for A and for B; thread covers chunks tid, tid+256
  const int ca0 = tid, ca1 = tid + 256;
  const __bf16* gA0 = A + (size_t)(m0 + (ca0 >> 2)) * K + 8 * (ca0 & 3);
  const __bf16* gA1 = A + (size_t)(m0 + (ca1 >> 2)) * K + 8 * (ca1 & 3);
  const __bf16* gB0 = B + (size_t)(ca0 >> 4) * N + n0 + 8 * (ca0 & 15);
  const __bf16* gB1 = B + (size_t)(ca1 >> 4) * N + n0 + 8 * (ca1 & 15);
  const uint32_t lA0 = (uint32_t)(uintptr_t)&As[0][(ca0 >> 2) * 32 + 8 * (ca0 & 3)];
  const uint32_t lA1 = (uint32_t)(uintptr_t)&As[0][(ca1 >> 2) * 32 + 8 * (ca1 & 3)];
  const uint32_t lB0 = (uint32_t)(uintptr_t)&Bs[0][(ca0 >> 4) * 128 + 8 * (ca0 & 15)];
  const uint32_t lB1 = (uint32_t)(uintptr_t)&Bs[0][(ca1 >> 4) * 128 + 8 * (ca1 & 15)];

  auto stage = [&](int buf, int ks) {
    const size_t koffA = (size_t)ks * 32;       // elements along K
    const size_t koffB = (size_t)ks * 32 * N;   // 32 rows of B
    const uint32_t bo = buf ? 8192u : 0u;       // bytes per LDS buffer
    async_ld16(lA0 + bo, gA0 + koffA);
    async_ld16(lA1 + bo, gA1 + koffA);
    async_ld16(lB0 + bo, gB0 + koffB);
    async_ld16(lB1 + bo, gB1 + koffB);
  };

  v8f acc[2][4];
#pragma unroll
  for (int t = 0; t < 2; ++t)
#pragma unroll
    for (int nt = 0; nt < 4; ++nt) { v8f z = {}; acc[t][nt] = z; }

  const int S = K >> 5;
  stage(0, 0);
  for (int s = 0; s < S; ++s) {
    wait_async0();        // own async writes for buffer s&1 complete
    __syncthreads();      // all waves' writes visible; prev reads finished
    if (s + 1 < S) stage((s + 1) & 1, s + 1);
    const int buf = s & 1;

    v16bf a[2];
#pragma unroll
    for (int t = 0; t < 2; ++t) {
      const int row = (wm * 32 + t * 16 + l16) * 32;
      v8bf lo = *(const v8bf*)&As[buf][row + 8 * half];
      v8bf hi = *(const v8bf*)&As[buf][row + 16 + 8 * half];
      a[t] = cat8(lo, hi);
    }
#pragma unroll
    for (int nt = 0; nt < 4; ++nt) {
      v16bf b;
#pragma unroll
      for (int e = 0; e < 16; ++e)
        b[e] = Bs[buf][(8 * half + (e & 7) + 16 * (e >> 3)) * 128 + wn * 64 + 16 * nt + l16];
      acc[0][nt] = __builtin_amdgcn_wmma_f32_16x16x32_bf16(false, a[0], false, b, (short)0, acc[0][nt], false, false);
      acc[1][nt] = __builtin_amdgcn_wmma_f32_16x16x32_bf16(false, a[1], false, b, (short)0, acc[1][nt], false, false);
    }
  }

#pragma unroll
  for (int t = 0; t < 2; ++t)
#pragma unroll
    for (int nt = 0; nt < 4; ++nt)
#pragma unroll
      for (int i = 0; i < 8; ++i) {
        const int r = m0 + wm * 32 + t * 16 + i + 8 * half;
        C[(size_t)r * N + n0 + wn * 64 + 16 * nt + l16] = acc[t][nt][i];
      }
}

// ---------------------------------------------------------------------------
// Elementwise f32 -> bf16
// ---------------------------------------------------------------------------
__global__ void f32_to_bf16_kernel(const float* __restrict__ in,
                                   __bf16* __restrict__ out, int n) {
  int i = blockIdx.x * blockDim.x + threadIdx.x;
  if (i < n) out[i] = (__bf16)in[i];
}

// in[R][Cc] f32 -> out[Cc][R] bf16 (output-coalesced transpose+convert)
__global__ void transpose_to_bf16_kernel(const float* __restrict__ in,
                                         __bf16* __restrict__ out, int R, int Cc) {
  int i = blockIdx.x * blockDim.x + threadIdx.x;
  if (i >= R * Cc) return;
  int c = i / R, r = i % R;
  out[(size_t)c * R + r] = (__bf16)in[(size_t)r * Cc + c];
}

// v slice of qkv -> bf16 [SEQ][KVSZ]
__global__ void v_to_bf16_kernel(const float* __restrict__ qkv,
                                 __bf16* __restrict__ vbf, int n) {
  int i = blockIdx.x * blockDim.x + threadIdx.x;
  if (i >= n) return;
  int t = i / KVSZ, jj = i % KVSZ;
  vbf[i] = (__bf16)qkv[(size_t)t * QKVN + QSZ + KVSZ + jj];
}

// ---------------------------------------------------------------------------
// RoPE (neox) + L2-norm (+ sqk^2 scaling for q).  One wave per (t, head).
// q written row-major bf16 [T][NH][D]; k written transposed [NKV][D][T].
// ---------------------------------------------------------------------------
__global__ __launch_bounds__(256) void rope_norm_kernel(
    const float* __restrict__ qkv, const int* __restrict__ positions,
    const float* __restrict__ sqk_w, __bf16* __restrict__ qbf,
    __bf16* __restrict__ kT) {
  const int lane = threadIdx.x & 31;
  const int wid  = (int)((blockIdx.x * blockDim.x + threadIdx.x) >> 5);
  const int t = wid / (NH + NKV);
  const int h = wid % (NH + NKV);
  if (t >= SEQ) return;
  const bool isq = (h < NH);
  const float* x = qkv + (size_t)t * QKVN + (isq ? h * DHEAD : QSZ + (h - NH) * DHEAD);
  const float pos = (float)positions[t];
  const float LN_THETA = 13.815510558f;  // ln(1e6)
  const int ia = lane, ib = lane + 32;
  const float fra = __expf(-((float)ia) * (LN_THETA / 64.f));
  const float frb = __expf(-((float)ib) * (LN_THETA / 64.f));
  float sa, ca, sb, cb;
  __sincosf(pos * fra, &sa, &ca);
  __sincosf(pos * frb, &sb, &cb);
  const float x1a = x[ia], x2a = x[ia + 64];
  const float x1b = x[ib], x2b = x[ib + 64];
  const float r1a = x1a * ca - x2a * sa;
  const float r2a = x1a * sa + x2a * ca;
  const float r1b = x1b * cb - x2b * sb;
  const float r2b = x1b * sb + x2b * cb;
  float ss = r1a * r1a + r2a * r2a + r1b * r1b + r2b * r2b;
#pragma unroll
  for (int m = 1; m < 32; m <<= 1) ss += __shfl_xor(ss, m, 32);
  const float inv = rsqrtf(ss);
  if (isq) {
    const float* sw = sqk_w + h * DHEAD;   // sqk = sqk_w * (1.0/0.02)
    const float s1a = sw[ia] * 50.f,      s2a = sw[ia + 64] * 50.f;
    const float s1b = sw[ib] * 50.f,      s2b = sw[ib + 64] * 50.f;
    __bf16* qr = qbf + ((size_t)t * NH + h) * DHEAD;
    qr[ia]      = (__bf16)(r1a * inv * s1a * s1a);
    qr[ia + 64] = (__bf16)(r2a * inv * s2a * s2a);
    qr[ib]      = (__bf16)(r1b * inv * s1b * s1b);
    qr[ib + 64] = (__bf16)(r2b * inv * s2b * s2b);
  } else {
    const int hk = h - NH;
    __bf16* kb = kT + (size_t)hk * DHEAD * SEQ;
    kb[(size_t)(ia)      * SEQ + t] = (__bf16)(r1a * inv);
    kb[(size_t)(ia + 64) * SEQ + t] = (__bf16)(r2a * inv);
    kb[(size_t)(ib)      * SEQ + t] = (__bf16)(r1b * inv);
    kb[(size_t)(ib + 64) * SEQ + t] = (__bf16)(r2b * inv);
  }
}

// ---------------------------------------------------------------------------
// Causal flash attention.  One wave per (head, 16-query tile); keys streamed
// in blocks of 32 up to the causal frontier.  Scores: 2x16-col WMMA tiles x
// 4 d-steps; online softmax with half-wave row reductions; P re-layouted
// through a per-wave LDS tile; 8 PV WMMAs accumulate the 16x128 output.
// ---------------------------------------------------------------------------
__global__ __launch_bounds__(256) void attn_kernel(
    const __bf16* __restrict__ qbf, const __bf16* __restrict__ kT,
    const __bf16* __restrict__ vbf, __bf16* __restrict__ obf) {
  __shared__ __align__(64) __bf16 Pld[8][16 * 32];
  const int lane = threadIdx.x & 31;
  const int wloc = threadIdx.x >> 5;
  const int wid  = blockIdx.x * 8 + wloc;
  const int h  = wid >> 7;          // SEQ/16 = 128 q-tiles per head
  const int qt = wid & 127;
  const int hk = h / GROUP;
  const int half = lane >> 4;
  const int l16  = lane & 15;
  const int q0 = qt * 16;
  const float NEGBIG = -3.402823e38f;
  const float scale = 11.313708499f;  // sqrt(D)

  v16bf Aq[4];
  const __bf16* qrow = qbf + ((size_t)(q0 + l16) * NH + h) * DHEAD;
#pragma unroll
  for (int s = 0; s < 4; ++s) {
    v8bf lo = *(const v8bf*)(qrow + 32 * s + 8 * half);
    v8bf hi = *(const v8bf*)(qrow + 32 * s + 16 + 8 * half);
    Aq[s] = cat8(lo, hi);
  }

  v8f O[8];
  float mrow[8], lrow[8], corr[8];
#pragma unroll
  for (int i = 0; i < 8; ++i) { v8f z = {}; O[i] = z; mrow[i] = NEGBIG; lrow[i] = 0.f; }

  for (int j = 0; j < q0 + 16; j += 32) {
    v8f S0 = {}, S1 = {};
#pragma unroll
    for (int s = 0; s < 4; ++s) {
      const __bf16* krow =
          kT + ((size_t)hk * DHEAD + 32 * s) * SEQ + j + l16;
      v16bf B0, B1;
#pragma unroll
      for (int e = 0; e < 16; ++e) {
        const size_t kk = (size_t)(8 * half + (e & 7) + 16 * (e >> 3)) * SEQ;
        B0[e] = krow[kk];
        B1[e] = krow[kk + 16];
      }
      S0 = __builtin_amdgcn_wmma_f32_16x16x32_bf16(false, Aq[s], false, B0, (short)0, S0, false, false);
      S1 = __builtin_amdgcn_wmma_f32_16x16x32_bf16(false, Aq[s], false, B1, (short)0, S1, false, false);
    }
#pragma unroll
    for (int i = 0; i < 8; ++i) {
      const int qg = q0 + i + 8 * half;
      float s0 = (j + l16      <= qg) ? S0[i] * scale : NEGBIG;
      float s1 = (j + 16 + l16 <= qg) ? S1[i] * scale : NEGBIG;
      float mx = fmaxf(s0, s1);
#pragma unroll
      for (int m = 1; m < 16; m <<= 1) mx = fmaxf(mx, __shfl_xor(mx, m, 32));
      const float mnew = fmaxf(mrow[i], mx);
      corr[i] = __expf(mrow[i] - mnew);
      const float e0 = __expf(s0 - mnew);
      const float e1 = __expf(s1 - mnew);
      float rs = e0 + e1;
#pragma unroll
      for (int m = 1; m < 16; m <<= 1) rs += __shfl_xor(rs, m, 32);
      lrow[i] = lrow[i] * corr[i] + rs;
      mrow[i] = mnew;
      const int r = i + 8 * half;
      Pld[wloc][r * 32 + l16]      = (__bf16)e0;
      Pld[wloc][r * 32 + 16 + l16] = (__bf16)e1;
    }
#pragma unroll
    for (int t = 0; t < 8; ++t)
#pragma unroll
      for (int i = 0; i < 8; ++i) O[t][i] *= corr[i];
    // same-wave DS ops are in order; compiler inserts s_wait_dscnt before use
    v8bf plo = *(const v8bf*)&Pld[wloc][l16 * 32 + 8 * half];
    v8bf phi = *(const v8bf*)&Pld[wloc][l16 * 32 + 16 + 8 * half];
    v16bf Pa = cat8(plo, phi);
#pragma unroll
    for (int t = 0; t < 8; ++t) {
      const __bf16* vrow =
          vbf + ((size_t)j * NKV + hk) * DHEAD + 16 * t + l16;
      v16bf Bv;
#pragma unroll
      for (int e = 0; e < 16; ++e)
        Bv[e] = vrow[(size_t)(8 * half + (e & 7) + 16 * (e >> 3)) * NKV * DHEAD];
      O[t] = __builtin_amdgcn_wmma_f32_16x16x32_bf16(false, Pa, false, Bv, (short)0, O[t], false, false);
    }
  }

#pragma unroll
  for (int t = 0; t < 8; ++t)
#pragma unroll
    for (int i = 0; i < 8; ++i)
      obf[(size_t)(q0 + i + 8 * half) * QSZ + h * DHEAD + 16 * t + l16] =
          (__bf16)(O[t][i] / lrow[i]);
}

// ---------------------------------------------------------------------------
extern "C" void kernel_launch(void* const* d_in, const int* in_sizes, int n_in,
                              void* d_out, int out_size, void* d_ws, size_t ws_size,
                              hipStream_t stream) {
  (void)in_sizes; (void)n_in; (void)out_size; (void)ws_size;
  const int*   positions = (const int*)d_in[0];
  const float* hidden    = (const float*)d_in[1];
  const float* w_qkv     = (const float*)d_in[2];
  const float* w_o       = (const float*)d_in[3];
  const float* sqk_w     = (const float*)d_in[4];
  float* out = (float*)d_out;
  char* ws = (char*)d_ws;

  auto al = [](size_t x) { return (x + 255) & ~(size_t)255; };
  const size_t oHbf = 0;
  const size_t oWq  = oHbf + al((size_t)SEQ * HIDSZ * 2);         // H bf16
  const size_t oWo  = oWq  + al((size_t)HIDSZ * QKVN * 2);        // WqkvT bf16
  const size_t oQKV = oWo  + al((size_t)QSZ * HIDSZ * 2);         // WoT bf16
  const size_t oQb  = oQKV + al((size_t)SEQ * QKVN * 4);          // qkv f32
  const size_t oKT  = oQb  + al((size_t)SEQ * QSZ * 2);           // q bf16
  const size_t oVb  = oKT  + al((size_t)NKV * DHEAD * SEQ * 2 + 256);  // kT (+pad)
  const size_t oOb  = oVb  + al((size_t)SEQ * KVSZ * 2 + 256);    // v (+pad)

  __bf16* Hbf   = (__bf16*)(ws + oHbf);
  __bf16* WqT   = (__bf16*)(ws + oWq);
  __bf16* WoT   = (__bf16*)(ws + oWo);
  float*  qkvf  = (float*)(ws + oQKV);
  __bf16* qbf   = (__bf16*)(ws + oQb);
  __bf16* kT    = (__bf16*)(ws + oKT);
  __bf16* vbf   = (__bf16*)(ws + oVb);
  __bf16* attnb = (__bf16*)(ws + oOb);

  // 1) convert H, transpose+convert weights
  {
    int n = SEQ * HIDSZ;
    f32_to_bf16_kernel<<<(n + 255) / 256, 256, 0, stream>>>(hidden, Hbf, n);
  }
  {
    int n = QKVN * HIDSZ;  // w_qkv [QKVN][HID] -> WqT [HID][QKVN]
    transpose_to_bf16_kernel<<<(n + 255) / 256, 256, 0, stream>>>(w_qkv, WqT, QKVN, HIDSZ);
  }
  {
    int n = HIDSZ * QSZ;   // w_o [HID][QSZ] -> WoT [QSZ][HID]
    transpose_to_bf16_kernel<<<(n + 255) / 256, 256, 0, stream>>>(w_o, WoT, HIDSZ, QSZ);
  }
  // 2) QKV GEMM: qkvf[SEQ][QKVN] = Hbf x WqT  (128x128 block tiles)
  {
    int blocks = (SEQ / 128) * (QKVN / 128);   // 16*48 = 768
    gemm_bf16_lds_kernel<<<blocks, 256, 0, stream>>>(Hbf, WqT, qkvf, SEQ, QKVN, HIDSZ);
  }
  // 3) RoPE + norm + scaling; v convert
  {
    int waves = SEQ * (NH + NKV);              // 81920
    rope_norm_kernel<<<waves * 32 / 256, 256, 0, stream>>>(qkvf, positions, sqk_w, qbf, kT);
  }
  {
    int n = SEQ * KVSZ;
    v_to_bf16_kernel<<<(n + 255) / 256, 256, 0, stream>>>(qkvf, vbf, n);
  }
  // 4) causal flash attention
  {
    int waves = NH * (SEQ / 16);               // 4096
    attn_kernel<<<waves / 8, 256, 0, stream>>>(qbf, kT, vbf, attnb);
  }
  // 5) output projection: out[SEQ][HID] = attnb x WoT
  {
    int blocks = (SEQ / 128) * (HIDSZ / 128);  // 16*32 = 512
    gemm_bf16_lds_kernel<<<blocks, 256, 0, stream>>>(attnb, WoT, out, SEQ, HIDSZ, QSZ);
  }
}
